// OmostDenseDiffusionCrossAttention_79439715106923
// MI455X (gfx1250) — compile-verified, matrix operand order checked by reading, not run on hardware
//
#include <hip/hip_runtime.h>
#include <math.h>

typedef __attribute__((ext_vector_type(16))) _Float16 v16h;
typedef __attribute__((ext_vector_type(8)))  _Float16 h8;
typedef __attribute__((ext_vector_type(4)))  _Float16 h4;
typedef __attribute__((ext_vector_type(8)))  float    v8f;

#define QLEN   4096
#define COND   77
#define KLEN   308
#define KPAD   320
#define NKT    20          // key tiles of 16
#define HEADS  8
#define DH     64
#define INNER  512
#define BATCH  2

#define KROW   72          // padded halfs per K row in LDS (144 B, 16B-aligned)
#define VSTR   328         // padded halfs per V^T / P row (656 B, 16B-aligned)
#define PSTR   328

#define KVBYTES (KPAD*KROW*2)      // 46080 B (also holds V^T: 64*328*2 = 41984 B)
#define PBYTES  (16*PSTR*2)        // 10496 B per wave
#define NWAVES  8
#define NTHREADS (NWAVES*32)
#define SMEM_BYTES (KVBYTES + NWAVES*PBYTES)   // 130048 B dynamic LDS

// -------- pre-kernel 1: per-region (mask_scale * 1/sqrt(dh)) ----------------
__global__ void dd_scales_kernel(const float* __restrict__ dd, float* __restrict__ scales) {
    __shared__ float red[256];
    const int rg = blockIdx.x;
    float s = 0.f;
    for (int i = threadIdx.x; i < 1024; i += 256) s += dd[rg * 1024 + i];
    red[threadIdx.x] = s;
    __syncthreads();
    for (int off = 128; off > 0; off >>= 1) {
        if ((int)threadIdx.x < off) red[threadIdx.x] += red[threadIdx.x + off];
        __syncthreads();
    }
    // mask_scale = 4096 / (4 * sum32x32) = 1024/sum ; fold in qk scale 0.125
    if (threadIdx.x == 0) scales[rg] = 128.0f / red[0];
}

// -------- pre-kernel 2: per-pixel 4-bit region coverage ---------------------
__global__ void dd_bits_kernel(const float* __restrict__ dd, unsigned* __restrict__ bits) {
    const int pix = blockIdx.x * 256 + threadIdx.x;
    if (pix >= QLEN) return;
    const int y = pix >> 6, x = pix & 63;
    const int mi = (y >> 1) * 32 + (x >> 1);   // nearest-exact 64->32
    unsigned b = 0;
#pragma unroll
    for (int rg = 0; rg < 4; ++rg)
        if (dd[rg * 1024 + mi] > 0.5f) b |= (1u << rg);
    bits[pix] = b;
}

// Build a 16-half WMMA fragment from two contiguous 16B runs in LDS.
__device__ __forceinline__ v16h frag_from_lds(const _Float16* base) {
    const h8 lo = *(const h8*)(base);
    const h8 hi = *(const h8*)(base + 16);
    return __builtin_shufflevector(lo, hi, 0, 1, 2, 3, 4, 5, 6, 7,
                                           8, 9, 10, 11, 12, 13, 14, 15);
}

// -------- main fused attention kernel ---------------------------------------
__global__ __launch_bounds__(NTHREADS, 1)
void dd_attn_kernel(const float* __restrict__ q, const float* __restrict__ k,
                    const float* __restrict__ v, const float* __restrict__ scales,
                    const unsigned* __restrict__ bits, float* __restrict__ out) {
    extern __shared__ _Float16 smem[];
    _Float16* kvbuf = smem;                                   // K, later V^T
    const int tid  = threadIdx.x;
    const int lane = tid & 31;
    const int wid  = tid >> 5;
    _Float16* pbuf = smem + (KVBYTES / 2) + wid * (PBYTES / 2);

    const int blk   = blockIdx.x;
    const int qblk  = blk & 31;                // 32 q-blocks of 128 rows
    const int h     = (blk >> 5) & 7;
    const int bi    = blk >> 8;
    const int qbase = qblk * 128 + wid * 16;   // this wave's 16 query rows

    const int l15 = lane & 15;
    const int grp = lane >> 4;

    const float* kg = k + (size_t)bi * KLEN * INNER + h * DH;
    const float* vg = v + (size_t)bi * KLEN * INNER + h * DH;

    // ---- stage K -> LDS f16 [KPAD][KROW]: float4 read + h4 (8B) DS store ---
    for (int idx = tid; idx < KPAD * (DH / 4); idx += NTHREADS) {
        const int key = idx >> 4, d = (idx & 15) * 4;
        h4 hv = (h4)(_Float16)0.f;
        if (key < KLEN) {
            const float4 f = *(const float4*)(kg + (size_t)key * INNER + d);
            hv[0] = (_Float16)f.x; hv[1] = (_Float16)f.y;
            hv[2] = (_Float16)f.z; hv[3] = (_Float16)f.w;
        }
        *(h4*)(kvbuf + key * KROW + d) = hv;
    }
    // prefetch V rows while we crunch phase 1
    for (int idx = tid; idx < KLEN; idx += NTHREADS)
        __builtin_prefetch(vg + (size_t)idx * INNER, 0, 1);
    __syncthreads();

    // ---- Q A-fragments (2 chunks of K=32): two 32B runs each ---------------
    const float* qg = q + ((size_t)bi * QLEN + qbase + l15) * INNER + h * DH;
    v16h aq[2];
#pragma unroll
    for (int c = 0; c < 2; ++c) {
#pragma unroll
        for (int half = 0; half < 2; ++half) {
            const int d = c * 32 + half * 16 + 8 * grp;
            const float4 f0 = *(const float4*)(qg + d);
            const float4 f1 = *(const float4*)(qg + d + 4);
            aq[c][half * 8 + 0] = (_Float16)f0.x; aq[c][half * 8 + 1] = (_Float16)f0.y;
            aq[c][half * 8 + 2] = (_Float16)f0.z; aq[c][half * 8 + 3] = (_Float16)f0.w;
            aq[c][half * 8 + 4] = (_Float16)f1.x; aq[c][half * 8 + 5] = (_Float16)f1.y;
            aq[c][half * 8 + 6] = (_Float16)f1.z; aq[c][half * 8 + 7] = (_Float16)f1.w;
        }
    }

    // ---- GEMM1: S[16 x 320] = Q K^T ----------------------------------------
    v8f S[NKT] = {};
#pragma unroll
    for (int t = 0; t < NKT; ++t) {
        const _Float16* kr = kvbuf + (t * 16 + l15) * KROW + 8 * grp;
#pragma unroll
        for (int c = 0; c < 2; ++c) {
            const v16h bf = frag_from_lds(kr + c * 32);
            S[t] = __builtin_amdgcn_wmma_f32_16x16x32_f16(
                false, aq[c], false, bf, (short)0, S[t], false, false);
        }
    }

    // ---- per-column metadata: scale + region bit (0 => always masked) ------
    float cs[NKT]; unsigned rgbit[NKT];
#pragma unroll
    for (int t = 0; t < NKT; ++t) {
        const int col = t * 16 + l15;
        int rg = col / COND; if (rg > 3) rg = 3;
        cs[t]    = scales[rg];                         // mask_scale * 1/sqrt(dh)
        rgbit[t] = (col < KLEN) ? (1u << rg) : 0u;     // 0 => key-pad column
    }
    unsigned rb[8];
#pragma unroll
    for (int r = 0; r < 8; ++r) rb[r] = bits[qbase + r + 8 * grp];

    // ---- masked softmax in C layout, write P (f16) to per-wave LDS ---------
#pragma unroll
    for (int r = 0; r < 8; ++r) {
        float m = -__builtin_inff();
#pragma unroll
        for (int t = 0; t < NKT; ++t) {
            const float x = S[t][r] * cs[t];
            const float a = ((rb[r] & rgbit[t]) != 0u) ? x : -__builtin_inff();
            S[t][r] = a;
            m = fmaxf(m, a);
        }
        m = fmaxf(m, __shfl_xor(m, 1, 32));
        m = fmaxf(m, __shfl_xor(m, 2, 32));
        m = fmaxf(m, __shfl_xor(m, 4, 32));
        m = fmaxf(m, __shfl_xor(m, 8, 32));
        float s = 0.f;
#pragma unroll
        for (int t = 0; t < NKT; ++t) {
            const float e = __expf(S[t][r] - m);
            S[t][r] = e;
            s += e;
        }
        s += __shfl_xor(s, 1, 32);
        s += __shfl_xor(s, 2, 32);
        s += __shfl_xor(s, 4, 32);
        s += __shfl_xor(s, 8, 32);
        const float inv = 1.f / s;
#pragma unroll
        for (int t = 0; t < NKT; ++t)
            pbuf[(r + 8 * grp) * PSTR + t * 16 + l15] = (_Float16)(S[t][r] * inv);
    }
    __syncthreads();   // all waves done reading K

    // ---- restage V transposed: VT[dim][key] f16, zero-pad keys >= 308 ------
    for (int idx = tid; idx < KPAD * (DH / 4); idx += NTHREADS) {
        const int key = idx >> 4, d = (idx & 15) * 4;
        _Float16 h0 = 0, h1 = 0, h2v = 0, h3 = 0;
        if (key < KLEN) {
            const float4 f = *(const float4*)(vg + (size_t)key * INNER + d);
            h0 = (_Float16)f.x; h1 = (_Float16)f.y;
            h2v = (_Float16)f.z; h3 = (_Float16)f.w;
        }
        kvbuf[(d + 0) * VSTR + key] = h0;
        kvbuf[(d + 1) * VSTR + key] = h1;
        kvbuf[(d + 2) * VSTR + key] = h2v;
        kvbuf[(d + 3) * VSTR + key] = h3;
    }
    __syncthreads();
    asm volatile("s_wait_dscnt 0" ::: "memory");

    // ---- GEMM2: O[16 x 64] = P V -------------------------------------------
    v8f O[4] = {};
    const _Float16* pr0 = pbuf + l15 * PSTR + 8 * grp;
#pragma unroll
    for (int c = 0; c < 10; ++c) {
        const v16h pf = frag_from_lds(pr0 + c * 32);
#pragma unroll
        for (int nt = 0; nt < 4; ++nt) {
            const _Float16* vr = kvbuf + (nt * 16 + l15) * VSTR + c * 32 + 8 * grp;
            const v16h bf = frag_from_lds(vr);
            O[nt] = __builtin_amdgcn_wmma_f32_16x16x32_f16(
                false, pf, false, bf, (short)0, O[nt], false, false);
        }
    }

    // ---- store O (f32, coalesced across l15) -------------------------------
#pragma unroll
    for (int nt = 0; nt < 4; ++nt)
#pragma unroll
        for (int r = 0; r < 8; ++r)
            out[((size_t)bi * QLEN + qbase + r + 8 * grp) * INNER + h * DH + nt * 16 + l15]
                = O[nt][r];
}

extern "C" void kernel_launch(void* const* d_in, const int* in_sizes, int n_in,
                              void* d_out, int out_size, void* d_ws, size_t ws_size,
                              hipStream_t stream) {
    const float* q  = (const float*)d_in[0];
    const float* k  = (const float*)d_in[1];
    const float* v  = (const float*)d_in[2];
    const float* dd = (const float*)d_in[3];
    // d_in[4] = heads scalar (== 8, baked into the kernel)
    float*    out    = (float*)d_out;
    float*    scales = (float*)d_ws;
    unsigned* bits   = (unsigned*)((char*)d_ws + 16);

    dd_scales_kernel<<<4, 256, 0, stream>>>(dd, scales);
    dd_bits_kernel<<<16, 256, 0, stream>>>(dd, bits);
    dd_attn_kernel<<<dim3(BATCH * HEADS * (QLEN / 128)), NTHREADS, SMEM_BYTES, stream>>>(
        q, k, v, scales, bits, out);
}
